// MultiHeadSelfAttention_35270271435131
// MI455X (gfx1250) — matrix operand order repaired and checked by reading, not compiled, not run on hardware
//
#include <hip/hip_runtime.h>
#include <hip/hip_bf16.h>

#define D_MODEL 1024
#define NUM_HEADS 16
#define DK 64
#define SEQ 2048
#define BATCH 4

typedef __attribute__((ext_vector_type(16))) __bf16 v16bf;
typedef __attribute__((ext_vector_type(8)))  float  v8f;

union Frag {
    uint4 q[2];
    v16bf v;
};

__device__ __forceinline__ unsigned short f32_bf16(float f) {
    unsigned int u = __builtin_bit_cast(unsigned int, f);
    unsigned int r = u + 0x7FFFu + ((u >> 16) & 1u);   // round-to-nearest-even
    return (unsigned short)(r >> 16);
}

__device__ __forceinline__ v8f wmma_bf16(const Frag& a, const Frag& b, v8f c) {
    // v_wmma_f32_16x16x32_bf16: D = A(16x32) * B(32x16) + C(16x16 f32)
    return __builtin_amdgcn_wmma_f32_16x16x32_bf16(
        /*neg_a=*/false, a.v, /*neg_b=*/false, b.v,
        /*c_mod=*/(short)0, c, /*reuse_a=*/false, /*reuse_b=*/false);
}

// ---------------------------------------------------------------------------
// Generic GEMM: C[M,N] = (A[M,K] * W[K,N] + bias[N]) * alpha
// Block tile 128x128, K-step 64 (2 WMMA k-subtiles per stage), 256 thr = 8
// waves, wave tile 64x32 -> 16 WMMAs per wave per barrier pair.
// ---------------------------------------------------------------------------
template<int A_BF16, int OUT_BF16>
__global__ __launch_bounds__(256)
void gemm_bias_kernel(const void* __restrict__ Ap, const float* __restrict__ W,
                      const float* __restrict__ bias, void* __restrict__ Cp,
                      int M, int N, int K, float alpha)
{
    // row stride 72 ushorts = 144 B (multiple of 16) -> b128-aligned fragments
    __shared__ unsigned short lA[128][72];   // [m][k] bf16
    __shared__ unsigned short lB[128][72];   // [n][k] bf16 (W transposed)

    const int tid  = threadIdx.x;
    const int m0g  = blockIdx.y * 128;
    const int n0g  = blockIdx.x * 128;
    const int wid  = tid >> 5;
    const int lane = tid & 31;
    const int lhalf = lane >> 4;
    const int l16   = lane & 15;
    const int waveM = (wid >> 2) * 64;
    const int waveN = (wid & 3) * 32;

    v8f acc[4][2];
#pragma unroll
    for (int mi = 0; mi < 4; ++mi)
#pragma unroll
        for (int ni = 0; ni < 2; ++ni)
            acc[mi][ni] = (v8f){0.f,0.f,0.f,0.f,0.f,0.f,0.f,0.f};

    for (int k0 = 0; k0 < K; k0 += 64) {
        __syncthreads();
        // ---- stage A tile: 128 rows x 64 k, 32 elements per thread ----
        {
            const int row = tid >> 1;
            const int kq  = (tid & 1) * 32;
            if (A_BF16) {
                const uint4* src = reinterpret_cast<const uint4*>(
                    (const unsigned short*)Ap + (size_t)(m0g + row) * K + k0 + kq);
                uint4 d0 = src[0], d1 = src[1], d2 = src[2], d3 = src[3];
                uint4* dst = reinterpret_cast<uint4*>(&lA[row][kq]);
                dst[0] = d0; dst[1] = d1; dst[2] = d2; dst[3] = d3;
            } else {
                const float4* src = reinterpret_cast<const float4*>(
                    (const float*)Ap + (size_t)(m0g + row) * K + k0 + kq);
                float4 f[8];
#pragma unroll
                for (int i = 0; i < 8; ++i) f[i] = src[i];
                const float* fv = reinterpret_cast<const float*>(f);
#pragma unroll
                for (int i = 0; i < 16; ++i) {
                    unsigned int u = (unsigned int)f32_bf16(fv[2*i]) |
                                     ((unsigned int)f32_bf16(fv[2*i+1]) << 16);
                    *reinterpret_cast<unsigned int*>(&lA[row][kq + 2*i]) = u;
                }
            }
        }
        // ---- stage W tile transposed: lB[n][k] = W[k0+k][n0g+n] ----
        {
            const int kk = tid >> 2;            // 0..63
            const int nq = (tid & 3) * 32;      // 0,32,64,96
            const float4* src = reinterpret_cast<const float4*>(
                W + (size_t)(k0 + kk) * N + n0g + nq);
            float4 g[8];
#pragma unroll
            for (int i = 0; i < 8; ++i) g[i] = src[i];
            const float* gv = reinterpret_cast<const float*>(g);
#pragma unroll
            for (int i = 0; i < 32; ++i)
                lB[nq + i][kk] = f32_bf16(gv[i]);
        }
        __syncthreads();

        // ---- 2 k-subtiles x 8 WMMAs per wave ----
#pragma unroll
        for (int ks = 0; ks < 64; ks += 32) {
            Frag af[4];
#pragma unroll
            for (int mi = 0; mi < 4; ++mi) {
                const unsigned short* rp = &lA[waveM + mi*16 + l16][0];
                af[mi].q[0] = *reinterpret_cast<const uint4*>(rp + ks + lhalf*8);
                af[mi].q[1] = *reinterpret_cast<const uint4*>(rp + ks + 16 + lhalf*8);
            }
            Frag bf[2];
#pragma unroll
            for (int ni = 0; ni < 2; ++ni) {
                const unsigned short* np = &lB[waveN + ni*16 + l16][0];
                bf[ni].q[0] = *reinterpret_cast<const uint4*>(np + ks + lhalf*8);
                bf[ni].q[1] = *reinterpret_cast<const uint4*>(np + ks + 16 + lhalf*8);
            }
#pragma unroll
            for (int mi = 0; mi < 4; ++mi)
#pragma unroll
                for (int ni = 0; ni < 2; ++ni)
                    acc[mi][ni] = wmma_bf16(af[mi], bf[ni], acc[mi][ni]);
        }
    }

    // ---- epilogue: (acc + bias) * alpha ----
#pragma unroll
    for (int ni = 0; ni < 2; ++ni) {
        const int ncol = n0g + waveN + ni*16 + l16;
        const float bval = bias[ncol];
#pragma unroll
        for (int mi = 0; mi < 4; ++mi) {
            const int rbase = m0g + waveM + mi*16 + lhalf*8;
#pragma unroll
            for (int j = 0; j < 8; ++j) {
                const float v = (acc[mi][ni][j] + bval) * alpha;
                const size_t idx = (size_t)(rbase + j) * N + ncol;
                if (OUT_BF16) ((unsigned short*)Cp)[idx] = f32_bf16(v);
                else          ((float*)Cp)[idx] = v;
            }
        }
    }
}

// ---------------------------------------------------------------------------
// Flash attention: block = (b, h, 128 q rows), 8 waves x 16 q rows each.
// Scores come pre-scaled by 1/sqrt(dk) (folded into Q projection).
// Row sums of P are accumulated on the matrix unit (P x ones) so only the
// row-max butterfly remains on ds_bpermute.
// ---------------------------------------------------------------------------
__global__ __launch_bounds__(256)
void attention_kernel(const unsigned short* __restrict__ Qw,
                      const unsigned short* __restrict__ Kw,
                      const unsigned short* __restrict__ Vw,
                      unsigned short* __restrict__ Ctx)
{
    __shared__ unsigned short lK[32][72];    // [key][dk]
    __shared__ unsigned short lVT[64][40];   // [dk][key]
    __shared__ unsigned short lP[8][16][32]; // per-wave P scratch

    const int tid  = threadIdx.x;
    const int wid  = tid >> 5;
    const int lane = tid & 31;
    const int lhalf = lane >> 4;
    const int l16   = lane & 15;
    const int bh = blockIdx.y;
    const int b  = bh >> 4;
    const int h  = bh & 15;
    const int qbase = blockIdx.x * 128 + wid * 16;
    const size_t headoff = (size_t)h * DK;

    // Q fragments (dk 0..31, 32..63) stay in registers
    Frag qf0, qf1;
    {
        const unsigned short* qp =
            Qw + (size_t)(b * SEQ + qbase + l16) * D_MODEL + headoff;
        qf0.q[0] = *reinterpret_cast<const uint4*>(qp + lhalf*8);
        qf0.q[1] = *reinterpret_cast<const uint4*>(qp + 16 + lhalf*8);
        qf1.q[0] = *reinterpret_cast<const uint4*>(qp + 32 + lhalf*8);
        qf1.q[1] = *reinterpret_cast<const uint4*>(qp + 48 + lhalf*8);
    }

    // all-ones bf16 B fragment for row-sum WMMA
    Frag onef;
    {
        const unsigned int one2 = 0x3F803F80u;   // two bf16 1.0
        onef.q[0] = make_uint4(one2, one2, one2, one2);
        onef.q[1] = make_uint4(one2, one2, one2, one2);
    }

    v8f ctx[4];
#pragma unroll
    for (int t = 0; t < 4; ++t)
        ctx[t] = (v8f){0.f,0.f,0.f,0.f,0.f,0.f,0.f,0.f};
    v8f rsum = (v8f){0.f,0.f,0.f,0.f,0.f,0.f,0.f,0.f};
    float rmax[8];
#pragma unroll
    for (int j = 0; j < 8; ++j) rmax[j] = -1e30f;

    for (int kt = 0; kt < SEQ; kt += 32) {
        __syncthreads();
        // ---- stage K[32][64] and V^T[64][32] ----
        {
            const int key = tid >> 3;
            const int dq  = (tid & 7) * 8;
            const size_t roff = (size_t)(b * SEQ + kt + key) * D_MODEL + headoff + dq;
            uint4 kd = *reinterpret_cast<const uint4*>(Kw + roff);
            *reinterpret_cast<uint4*>(&lK[key][dq]) = kd;
            uint4 vd = *reinterpret_cast<const uint4*>(Vw + roff);
            const unsigned short* vs = reinterpret_cast<const unsigned short*>(&vd);
#pragma unroll
            for (int i = 0; i < 8; ++i) lVT[dq + i][key] = vs[i];
        }
        __syncthreads();

        // ---- scores: two 16x16 tiles, dk=64 => 2 k-steps each ----
        v8f s0 = (v8f){0.f,0.f,0.f,0.f,0.f,0.f,0.f,0.f};
        v8f s1 = s0;
        {
            Frag kf;
            const unsigned short* kp0 = &lK[l16][0];
            kf.q[0] = *reinterpret_cast<const uint4*>(kp0 + lhalf*8);
            kf.q[1] = *reinterpret_cast<const uint4*>(kp0 + 16 + lhalf*8);
            s0 = wmma_bf16(qf0, kf, s0);
            kf.q[0] = *reinterpret_cast<const uint4*>(kp0 + 32 + lhalf*8);
            kf.q[1] = *reinterpret_cast<const uint4*>(kp0 + 48 + lhalf*8);
            s0 = wmma_bf16(qf1, kf, s0);
            const unsigned short* kp1 = &lK[16 + l16][0];
            kf.q[0] = *reinterpret_cast<const uint4*>(kp1 + lhalf*8);
            kf.q[1] = *reinterpret_cast<const uint4*>(kp1 + 16 + lhalf*8);
            s1 = wmma_bf16(qf0, kf, s1);
            kf.q[0] = *reinterpret_cast<const uint4*>(kp1 + 32 + lhalf*8);
            kf.q[1] = *reinterpret_cast<const uint4*>(kp1 + 48 + lhalf*8);
            s1 = wmma_bf16(qf1, kf, s1);
        }

        // ---- online softmax: row max via butterfly, rescale all accs ----
#pragma unroll
        for (int j = 0; j < 8; ++j) {
            float t = fmaxf(s0[j], s1[j]);
#pragma unroll
            for (int off = 1; off < 16; off <<= 1)
                t = fmaxf(t, __shfl_xor(t, off, 32));
            const float m_new = fmaxf(rmax[j], t);
            const float scale = __expf(rmax[j] - m_new);
            rmax[j] = m_new;
            s0[j] = __expf(s0[j] - m_new);
            s1[j] = __expf(s1[j] - m_new);
            rsum[j] *= scale;
#pragma unroll
            for (int t4 = 0; t4 < 4; ++t4) ctx[t4][j] *= scale;
        }

        // ---- pack P (16x32 bf16) through per-wave LDS into an A fragment ----
        {
            unsigned short* pw = &lP[wid][0][0];
#pragma unroll
            for (int j = 0; j < 8; ++j) {
                const int r = j + lhalf * 8;
                pw[r*32 + l16]      = f32_bf16(s0[j]);
                pw[r*32 + 16 + l16] = f32_bf16(s1[j]);
            }
        }
#if __has_builtin(__builtin_amdgcn_s_wait_dscnt)
        __builtin_amdgcn_s_wait_dscnt(0);
#else
        asm volatile("s_wait_dscnt 0" ::: "memory");
#endif
        Frag pf;
        {
            const unsigned short* pr = &lP[wid][l16][0];
            pf.q[0] = *reinterpret_cast<const uint4*>(pr + lhalf*8);
            pf.q[1] = *reinterpret_cast<const uint4*>(pr + 16 + lhalf*8);
        }

        // ---- rowsum += P * ones (on the matrix unit) ----
        rsum = wmma_bf16(pf, onef, rsum);

        // ---- ctx += P * V : 4 n-tiles over dk ----
#pragma unroll
        for (int t = 0; t < 4; ++t) {
            const unsigned short* vr = &lVT[t*16 + l16][0];
            Frag vf;
            vf.q[0] = *reinterpret_cast<const uint4*>(vr + lhalf*8);
            vf.q[1] = *reinterpret_cast<const uint4*>(vr + 16 + lhalf*8);
            ctx[t] = wmma_bf16(pf, vf, ctx[t]);
        }
    }

    // ---- normalize and store ctx (bf16, [B*S, D_MODEL] layout) ----
#pragma unroll
    for (int t = 0; t < 4; ++t) {
        const int dcol = t*16 + l16;
#pragma unroll
        for (int j = 0; j < 8; ++j) {
            const int qrow = qbase + lhalf*8 + j;
            const float v = ctx[t][j] / rsum[j];
            Ctx[(size_t)(b * SEQ + qrow) * D_MODEL + headoff + dcol] = f32_bf16(v);
        }
    }
}

// ---------------------------------------------------------------------------
extern "C" void kernel_launch(void* const* d_in, const int* in_sizes, int n_in,
                              void* d_out, int out_size, void* d_ws, size_t ws_size,
                              hipStream_t stream)
{
    const float* x  = (const float*)d_in[0];
    const float* Wq = (const float*)d_in[1];
    const float* bq = (const float*)d_in[2];
    const float* Wk = (const float*)d_in[3];
    const float* bk = (const float*)d_in[4];
    const float* Wv = (const float*)d_in[5];
    const float* bv = (const float*)d_in[6];
    const float* Wo = (const float*)d_in[7];
    const float* bo = (const float*)d_in[8];
    float* out = (float*)d_out;

    const int M = BATCH * SEQ;               // 8192
    const size_t mat = (size_t)M * D_MODEL;  // 8M elems
    unsigned short* Qw = (unsigned short*)d_ws;
    unsigned short* Kw = Qw + mat;
    unsigned short* Vw = Kw + mat;
    unsigned short* Cw = Vw + mat;           // 64 MB total bf16 workspace

    dim3 blk(256);
    dim3 gg(D_MODEL / 128, M / 128);

    const float qscale = 0.125f;             // 1/sqrt(dk), folded into Q
    gemm_bias_kernel<0,1><<<gg, blk, 0, stream>>>(x, Wq, bq, Qw, M, D_MODEL, D_MODEL, qscale);
    gemm_bias_kernel<0,1><<<gg, blk, 0, stream>>>(x, Wk, bk, Kw, M, D_MODEL, D_MODEL, 1.0f);
    gemm_bias_kernel<0,1><<<gg, blk, 0, stream>>>(x, Wv, bv, Vw, M, D_MODEL, D_MODEL, 1.0f);

    attention_kernel<<<dim3(SEQ / 128, BATCH * NUM_HEADS), blk, 0, stream>>>(Qw, Kw, Vw, Cw);

    gemm_bias_kernel<1,0><<<gg, blk, 0, stream>>>(Cw, Wo, bo, out, M, D_MODEL, D_MODEL, 1.0f);
}
